// DetNet_38757784879682
// MI455X (gfx1250) — compile-verified, matrix operand order checked by reading, not run on hardware
//
#include <hip/hip_runtime.h>
#include <hip/hip_bf16.h>

// ---------------- problem constants (from reference) ----------------
#define TSTEPS 50
#define NA 23
#define BS 64
#define NTOT (NA * BS)      // 1472
#define XD 2
#define HD 64
#define RD 64
#define GH 16
#define NH 4
#define GD 9
#define ALPHA_GAT 0.2f

typedef __attribute__((ext_vector_type(16))) _Float16 v16h;
typedef __attribute__((ext_vector_type(8)))  _Float16 v8h;
typedef __attribute__((ext_vector_type(8)))  float    v8f;

// ---------------- WMMA helpers (gfx1250 wave32 layouts) ----------------
// A-matrix 16x32 f16 fragment. lane<16: row=lane, K = {kb..kb+7, kb+16..kb+23}
// with kb=kb0; lane>=16: row=lane-16, kb=kb0+8.  (ISA 7.12.2)
// All call sites guarantee 16-byte alignment: lda multiple of 8, kb multiple of 8.
__device__ __forceinline__ v16h frag_A(const _Float16* A, int lda, int lane, int kb0) {
    int row = lane & 15;
    int kb  = kb0 + ((lane & 16) >> 1);      // +8 for upper half-wave
    const _Float16* p = A + (size_t)row * lda + kb;
    v8h lo = *reinterpret_cast<const v8h*>(p);
    v8h hi = *reinterpret_cast<const v8h*>(p + 16);
    v16h a;
#pragma unroll
    for (int e = 0; e < 8; ++e) a[e] = lo[e];
#pragma unroll
    for (int e = 0; e < 8; ++e) a[8 + e] = hi[e];
    return a;
}

// B-matrix fragment from pre-packed (fragment-major) weight buffer:
// contiguous 32 bytes per lane -> fully coalesced 1KB per wave.
__device__ __forceinline__ v16h frag_Bp(const _Float16* Bp, int lane) {
    return *reinterpret_cast<const v16h*>(Bp + (lane << 4));
}

__device__ __forceinline__ v8f wmma_f16(v16h a, v16h b, v8f c) {
    return __builtin_amdgcn_wmma_f32_16x16x32_f16(false, a, false, b,
                                                  (short)0, c, false, false);
}

__device__ __forceinline__ float sigm(float x) { return 1.f / (1.f + __expf(-x)); }
__device__ __forceinline__ float leaky(float x, float s) { return x >= 0.f ? x : s * x; }
__device__ __forceinline__ float elu1(float x) { return x > 0.f ? x : __expf(x) - 1.f; }

// Decode fragment-major pack index -> (ct, matrix col, k)
__device__ __forceinline__ void pack_decode(int i, int numKC, int& ct, int& col, int& k) {
    int e    = i & 15;
    int lane = (i >> 4) & 31;
    int rest = i >> 9;
    int kc   = rest % numKC;
    ct       = rest / numKC;
    col      = (ct << 4) + (lane & 15);
    k        = kc * 32 + ((lane & 16) >> 1) + (e < 8 ? e : e + 8);
}

// ---------------- prep: repack weights to f16 fragment-major, zero state ----------------
__global__ void k_prep(const float* __restrict__ w2, const float* __restrict__ wih,
                       const float* __restrict__ whh, const float* __restrict__ gW,
                       const float* __restrict__ Wo, const float* __restrict__ lgw,
                       _Float16* Bphi2, _Float16* Bwih, _Float16* Bwhh,
                       _Float16* BgatW, _Float16* BWo, _Float16* Blg,
                       float* h, _Float16* h16, float* loss) {
    int i0 = blockIdx.x * blockDim.x + threadIdx.x;
    int stride = gridDim.x * blockDim.x;
    if (i0 == 0) loss[0] = 0.f;
    for (int i = i0; i < NTOT * RD; i += stride) { h[i] = 0.f; h16[i] = (_Float16)0.f; }
    // 64x64 matrices (numKC = 2)
    for (int i = i0; i < 64 * 64; i += stride) {
        int ct, col, k; pack_decode(i, 2, ct, col, k);
        Bphi2[i] = (_Float16)w2[col * 64 + k];                        // phi_w2^T
        BgatW[i] = (_Float16)gW[(col >> 4) * 1024 + k * 16 + (col & 15)]; // 'hdf'
        BWo[i]   = (_Float16)Wo[k * 64 + col];                        // K-major src
    }
    // 64x192 GRU weights (C=192, numKC = 2)
    for (int i = i0; i < 64 * 192; i += stride) {
        int ct, col, k; pack_decode(i, 2, ct, col, k);
        Bwih[i] = (_Float16)wih[col * 64 + k];                        // gru_wih^T
        Bwhh[i] = (_Float16)whh[col * 64 + k];                        // gru_whh^T
    }
    // 128x64 output projection (numKC = 4)
    for (int i = i0; i < 128 * 64; i += stride) {
        int ct, col, k; pack_decode(i, 4, ct, col, k);
        Blg[i] = (_Float16)lgw[col * 128 + k];                        // lg_w^T
    }
}

// ---------------- fused phi-MLP + GRU step ----------------
// grid 23 x 256 threads (8 waves); each workgroup owns 64 rows.
__global__ void __launch_bounds__(256)
k_gru(const float* __restrict__ xt, const float* __restrict__ w1,
      const float* __restrict__ b1, const _Float16* __restrict__ Bphi2,
      const float* __restrict__ b2, const _Float16* __restrict__ Bwih,
      const _Float16* __restrict__ Bwhh, const float* __restrict__ bih,
      const float* __restrict__ bhh, const float* __restrict__ hf32,
      const _Float16* __restrict__ h16, _Float16* __restrict__ cat16) {
    __shared__ _Float16 sA1[64 * 64];   // phi layer1 output (f16)
    __shared__ _Float16 sPhi[64 * 64];  // phi layer2 output (f16)

    const int tid  = threadIdx.x;
    const int lane = tid & 31;
    const int w    = tid >> 5;
    const int r0   = blockIdx.x * 64;

    // --- phi layer1 (K=2, scalar) ---
    for (int i = tid; i < 64 * 64; i += 256) {
        int nl = i >> 6, j = i & 63;
        float x0 = xt[(size_t)(r0 + nl) * XD + 0];
        float x1 = xt[(size_t)(r0 + nl) * XD + 1];
        float v = x0 * w1[j * XD + 0] + x1 * w1[j * XD + 1] + b1[j];
        sA1[nl * 64 + j] = (_Float16)leaky(v, 0.01f);
    }
    __syncthreads();

    // --- phi layer2 (64x64 @ 64x64, WMMA) ---
    {
        const int col = lane & 15, hi = (lane >> 4) * 8;
#pragma unroll
        for (int tt = 0; tt < 2; ++tt) {
            int t = w + tt * 8;             // tiles 0..15
            int tr = t >> 2, tc = t & 3;
            v8f acc = {0.f, 0.f, 0.f, 0.f, 0.f, 0.f, 0.f, 0.f};
#pragma unroll
            for (int kc = 0; kc < 2; ++kc) {
                v16h a = frag_A(sA1 + tr * 16 * 64, 64, lane, kc * 32);
                v16h b = frag_Bp(Bphi2 + ((tc * 2 + kc) << 9), lane);
                acc = wmma_f16(a, b, acc);
            }
            int jj = tc * 16 + col;
            float bb = b2[jj];
#pragma unroll
            for (int v = 0; v < 8; ++v) {
                int r = tr * 16 + v + hi;
                sPhi[r * 64 + jj] = (_Float16)leaky(acc[v] + bb, 0.01f);
            }
        }
    }
    __syncthreads();

    // --- GRU gates: gi = phi@wih^T, gh = hprev@whh^T, all in registers ---
    {
        const int col = lane & 15, hi = (lane >> 4) * 8;
        const int tr = w & 3;               // row tile
        const int half = w >> 2;            // col-tile pair selector
#pragma unroll
        for (int tci = 0; tci < 2; ++tci) {
            int tc = half * 2 + tci;        // 0..3 (within 64-wide gate slab)
            v8f ai0 = {0,0,0,0,0,0,0,0}, ai1 = {0,0,0,0,0,0,0,0}, ai2 = {0,0,0,0,0,0,0,0};
            v8f ah0 = {0,0,0,0,0,0,0,0}, ah1 = {0,0,0,0,0,0,0,0}, ah2 = {0,0,0,0,0,0,0,0};
#pragma unroll
            for (int kc = 0; kc < 2; ++kc) {
                v16h aP = frag_A(sPhi + tr * 16 * 64, 64, lane, kc * 32);
                v16h aH = frag_A(h16 + (size_t)(r0 + tr * 16) * 64, 64, lane, kc * 32);
                // B col-tile index within 192-wide matrix: ct = g*4 + tc
                v16h bi0 = frag_Bp(Bwih + (((0 * 4 + tc) * 2 + kc) << 9), lane);
                v16h bi1 = frag_Bp(Bwih + (((1 * 4 + tc) * 2 + kc) << 9), lane);
                v16h bi2 = frag_Bp(Bwih + (((2 * 4 + tc) * 2 + kc) << 9), lane);
                v16h bh0 = frag_Bp(Bwhh + (((0 * 4 + tc) * 2 + kc) << 9), lane);
                v16h bh1 = frag_Bp(Bwhh + (((1 * 4 + tc) * 2 + kc) << 9), lane);
                v16h bh2 = frag_Bp(Bwhh + (((2 * 4 + tc) * 2 + kc) << 9), lane);
                ai0 = wmma_f16(aP, bi0, ai0);
                ai1 = wmma_f16(aP, bi1, ai1);
                ai2 = wmma_f16(aP, bi2, ai2);
                ah0 = wmma_f16(aH, bh0, ah0);
                ah1 = wmma_f16(aH, bh1, ah1);
                ah2 = wmma_f16(aH, bh2, ah2);
            }
            int jj = tc * 16 + col;
            float bir = bih[jj], biz = bih[64 + jj], bin = bih[128 + jj];
            float bhr = bhh[jj], bhz = bhh[64 + jj], bhn = bhh[128 + jj];
#pragma unroll
            for (int v = 0; v < 8; ++v) {
                int row = r0 + tr * 16 + v + hi;
                float r = sigm((ai0[v] + bir) + (ah0[v] + bhr));
                float z = sigm((ai1[v] + biz) + (ah1[v] + bhz));
                float nn = tanhf((ai2[v] + bin) + r * (ah2[v] + bhn));
                float hp = hf32[(size_t)row * 64 + jj];
                float h1 = (1.f - z) * nn + z * hp;
                cat16[(size_t)row * 128 + 64 + jj] = (_Float16)h1;  // h1 -> cat cols 64..127
            }
        }
    }
}

// ---------------- generic N x 64 GEMM (out = A[N x K] @ B[K x 64] (+bias)) ----------------
// grid 23 x 256; A f16 row-major (lda f16 elems), B pre-packed fragment-major.
__global__ void __launch_bounds__(256)
k_gemm64(const _Float16* __restrict__ A, int lda, const _Float16* __restrict__ B,
         int K, const float* __restrict__ bias, float* __restrict__ Cf32,
         _Float16* __restrict__ Cf16) {
    __shared__ _Float16 sA[64 * 128];
    const int tid  = threadIdx.x;
    const int lane = tid & 31;
    const int w    = tid >> 5;
    const int r0   = blockIdx.x * 64;

    // stage A tile (64 x K) into LDS, 4 bytes at a time
    const uint32_t* Au = reinterpret_cast<const uint32_t*>(A);
    const int kh = K >> 1;                   // uints per row
    for (int i = tid; i < 64 * kh; i += 256) {
        int row = i / kh, c = i - row * kh;
        reinterpret_cast<uint32_t*>(sA)[row * kh + c] =
            Au[(size_t)(r0 + row) * (lda >> 1) + c];
    }
    __syncthreads();

    const int col = lane & 15, hi = (lane >> 4) * 8;
    const int nkc = K >> 5;
#pragma unroll
    for (int tt = 0; tt < 2; ++tt) {
        int t = w + tt * 8;
        int tr = t >> 2, tc = t & 3;
        v8f acc = {0.f, 0.f, 0.f, 0.f, 0.f, 0.f, 0.f, 0.f};
        for (int kc = 0; kc < nkc; ++kc) {
            v16h a = frag_A(sA + tr * 16 * K, K, lane, kc * 32);
            v16h b = frag_Bp(B + ((tc * nkc + kc) << 9), lane);
            acc = wmma_f16(a, b, acc);
        }
        int jj = tc * 16 + col;
        float bb = bias ? bias[jj] : 0.f;
#pragma unroll
        for (int v = 0; v < 8; ++v) {
            int row = r0 + tr * 16 + v + hi;
            float val = acc[v] + bb;
            if (Cf32) Cf32[(size_t)row * 64 + jj] = val;
            if (Cf16) Cf16[(size_t)row * 64 + jj] = (_Float16)val;
        }
    }
}

// ---------------- GAT stage 1: per-block 4-head attention on Wh ----------------
// grid 64 (one adjacency block of 23 agents), 128 threads.
__global__ void __launch_bounds__(128)
k_attn1(const float* __restrict__ Wh, const float* __restrict__ ga,
        _Float16* __restrict__ xcat16) {
    __shared__ float sW[NA * 64];
    __shared__ float sFi[NH * NA], sFj[NH * NA];
    const int base = blockIdx.x * NA;
    for (int i = threadIdx.x; i < NA * 64; i += 128)
        sW[i] = Wh[(size_t)base * 64 + i];
    __syncthreads();
    int t = threadIdx.x;
    if (t < NH * NA) {
        int hh = t / NA, n = t - hh * NA;
        float fi = 0.f, fj = 0.f;
#pragma unroll
        for (int f = 0; f < GH; ++f) {
            float v = sW[n * 64 + hh * GH + f];
            fi += v * ga[hh * 2 * GH + f];
            fj += v * ga[hh * 2 * GH + GH + f];
        }
        sFi[t] = fi; sFj[t] = fj;
    }
    __syncthreads();
    if (t < NH * NA) {
        int hh = t / NA, n = t - hh * NA;
        float fi = sFi[hh * NA + n];
        float p[NA], mx = -1e30f;
#pragma unroll
        for (int m = 0; m < NA; ++m) {
            float e = leaky(fi + sFj[hh * NA + m], ALPHA_GAT);
            p[m] = e; mx = fmaxf(mx, e);
        }
        float s = 0.f;
#pragma unroll
        for (int m = 0; m < NA; ++m) { p[m] = __expf(p[m] - mx); s += p[m]; }
        float inv = 1.f / s;
#pragma unroll
        for (int f = 0; f < GH; ++f) {
            float acc = 0.f;
#pragma unroll
            for (int m = 0; m < NA; ++m) acc += p[m] * sW[m * 64 + hh * GH + f];
            acc = elu1(acc * inv);
            xcat16[(size_t)(base + n) * 64 + hh * GH + f] = (_Float16)acc;
        }
    }
}

// ---------------- GAT stage 2: per-block single-head (F=64) attention ----------------
__global__ void __launch_bounds__(128)
k_attn2(const float* __restrict__ Who, const float* __restrict__ ao,
        _Float16* __restrict__ cat16) {
    __shared__ float sW[NA * 64];
    __shared__ float sFi[NA], sFj[NA];
    __shared__ float sP[NA * NA];
    const int base = blockIdx.x * NA;
    for (int i = threadIdx.x; i < NA * 64; i += 128)
        sW[i] = Who[(size_t)base * 64 + i];
    __syncthreads();
    int t = threadIdx.x;
    if (t < NA) {
        float fi = 0.f, fj = 0.f;
#pragma unroll
        for (int f = 0; f < 64; ++f) {
            float v = sW[t * 64 + f];
            fi += v * ao[f];
            fj += v * ao[64 + f];
        }
        sFi[t] = fi; sFj[t] = fj;
    }
    __syncthreads();
    if (t < NA) {
        float fi = sFi[t];
        float p[NA], mx = -1e30f;
#pragma unroll
        for (int m = 0; m < NA; ++m) {
            float e = leaky(fi + sFj[m], ALPHA_GAT);
            p[m] = e; mx = fmaxf(mx, e);
        }
        float s = 0.f;
#pragma unroll
        for (int m = 0; m < NA; ++m) { p[m] = __expf(p[m] - mx); s += p[m]; }
        float inv = 1.f / s;
#pragma unroll
        for (int m = 0; m < NA; ++m) sP[t * NA + m] = p[m] * inv;
    }
    __syncthreads();
    for (int i = threadIdx.x; i < NA * 64; i += 128) {
        int n = i >> 6, f = i & 63;
        float acc = 0.f;
#pragma unroll
        for (int m = 0; m < NA; ++m) acc += sP[n * NA + m] * sW[m * 64 + f];
        cat16[(size_t)(base + n) * 128 + f] = (_Float16)elu1(acc); // h_ref -> cols 0..63
    }
}

// ---------------- classifier + NLL on ball rows (every 23rd row) ----------------
__global__ void __launch_bounds__(64)
k_loss(const float* __restrict__ h, const float* __restrict__ cw,
       const float* __restrict__ cb, const float* __restrict__ tgt,
       float* __restrict__ loss) {
    __shared__ float sl[BS];
    int b = threadIdx.x;                 // 0..63
    int row = b * NA;
    float lg[GD], mx = -1e30f;
#pragma unroll
    for (int g = 0; g < GD; ++g) {
        float a = cb[g];
#pragma unroll
        for (int k = 0; k < 64; ++k) a += h[(size_t)row * 64 + k] * cw[g * 64 + k];
        lg[g] = a; mx = fmaxf(mx, a);
    }
    float s = 0.f;
#pragma unroll
    for (int g = 0; g < GD; ++g) { lg[g] -= mx; s += __expf(lg[g]); }
    float ls = __logf(s);
    float lp = 0.f;
#pragma unroll
    for (int g = 0; g < GD; ++g) lp += tgt[b * GD + g] * (lg[g] - ls);
    sl[b] = lp;
    __syncthreads();
    if (b == 0) {
        float sum = 0.f;
#pragma unroll
        for (int i = 0; i < BS; ++i) sum += sl[i];
        loss[0] -= sum * (1.f / (float)BS);   // loss = -sum_t mean(logp)
    }
}

// ---------------- final: copy hT into d_out[1..] ----------------
__global__ void k_final(const float* __restrict__ h, float* __restrict__ out) {
    int i = blockIdx.x * blockDim.x + threadIdx.x;
    if (i < NTOT * RD) out[1 + i] = h[i];
}

// ---------------- host launcher ----------------
extern "C" void kernel_launch(void* const* d_in, const int* in_sizes, int n_in,
                              void* d_out, int out_size, void* d_ws, size_t ws_size,
                              hipStream_t stream) {
    const float* traj = (const float*)d_in[0];   // (T, N, 2)
    const float* tgts = (const float*)d_in[1];   // (T, 64, 9)
    // d_in[2] = adj (ignored: block structure is static), d_in[3] = n_agents
    const float* phi_w1 = (const float*)d_in[4];
    const float* phi_b1 = (const float*)d_in[5];
    const float* phi_w2 = (const float*)d_in[6];
    const float* phi_b2 = (const float*)d_in[7];
    const float* gru_wih = (const float*)d_in[8];
    const float* gru_whh = (const float*)d_in[9];
    const float* gru_bih = (const float*)d_in[10];
    const float* gru_bhh = (const float*)d_in[11];
    const float* gat_W  = (const float*)d_in[12];
    const float* gat_a  = (const float*)d_in[13];
    const float* gat_Wo = (const float*)d_in[14];
    const float* gat_ao = (const float*)d_in[15];
    const float* lg_w   = (const float*)d_in[16];
    const float* lg_b   = (const float*)d_in[17];
    const float* cls_w  = (const float*)d_in[18];
    const float* cls_b  = (const float*)d_in[19];

    char* ws = (char*)d_ws;
    size_t off = 0;
    auto alloc = [&](size_t bytes) { char* p = ws + off; off += (bytes + 255) & ~size_t(255); return p; };
    float*     h_f32  = (float*)    alloc((size_t)NTOT * 64 * 4);
    _Float16*  h16    = (_Float16*) alloc((size_t)NTOT * 64 * 2);
    _Float16*  cat16  = (_Float16*) alloc((size_t)NTOT * 128 * 2);
    float*     Wh     = (float*)    alloc((size_t)NTOT * 64 * 4);
    _Float16*  xcat16 = (_Float16*) alloc((size_t)NTOT * 64 * 2);
    float*     Who    = (float*)    alloc((size_t)NTOT * 64 * 4);
    _Float16*  Bphi2  = (_Float16*) alloc(64 * 64 * 2);
    _Float16*  Bwih   = (_Float16*) alloc(64 * 192 * 2);
    _Float16*  Bwhh   = (_Float16*) alloc(64 * 192 * 2);
    _Float16*  BgatW  = (_Float16*) alloc(64 * 64 * 2);
    _Float16*  BWo    = (_Float16*) alloc(64 * 64 * 2);
    _Float16*  Blg    = (_Float16*) alloc(128 * 64 * 2);

    float* loss = (float*)d_out;

    k_prep<<<128, 256, 0, stream>>>(phi_w2, gru_wih, gru_whh, gat_W, gat_Wo, lg_w,
                                    Bphi2, Bwih, Bwhh, BgatW, BWo, Blg,
                                    h_f32, h16, loss);

    for (int t = 1; t < TSTEPS; ++t) {
        const float* xt  = traj + (size_t)t * NTOT * XD;
        const float* gt  = tgts + (size_t)t * BS * GD;
        k_gru<<<NTOT / 64, 256, 0, stream>>>(xt, phi_w1, phi_b1, Bphi2, phi_b2,
                                             Bwih, Bwhh, gru_bih, gru_bhh,
                                             h_f32, h16, cat16);
        // Wh = h1 @ gat_W  (h1 lives in cat16 cols 64..127)
        k_gemm64<<<NTOT / 64, 256, 0, stream>>>(cat16 + 64, 128, BgatW, 64,
                                                nullptr, Wh, nullptr);
        k_attn1<<<BS, 128, 0, stream>>>(Wh, gat_a, xcat16);
        // Who = xcat @ gat_Wo
        k_gemm64<<<NTOT / 64, 256, 0, stream>>>(xcat16, 64, BWo, 64,
                                                nullptr, Who, nullptr);
        k_attn2<<<BS, 128, 0, stream>>>(Who, gat_ao, cat16);
        // h_new = [h_ref | h1] @ lg_w^T + lg_b  -> overwrites carry (f32 + f16)
        k_gemm64<<<NTOT / 64, 256, 0, stream>>>(cat16, 128, Blg, 128,
                                                lg_b, h_f32, h16);
        k_loss<<<1, BS, 0, stream>>>(h_f32, cls_w, cls_b, gt, loss);
    }
    k_final<<<(NTOT * RD + 255) / 256, 256, 0, stream>>>(h_f32, (float*)d_out);
    (void)in_sizes; (void)n_in; (void)out_size; (void)ws_size;
}
